// RNNTJoint_47467978555668
// MI455X (gfx1250) — compile-verified, matrix operand order checked by reading, not run on hardware
//
#include <hip/hip_runtime.h>

// ---- problem constants (from reference) ----
#define BB   4
#define TT   256
#define UU   96
#define EH   1024
#define PH   640
#define JH   640
#define VV   1025
#define VPAD 1040   // 65 * 16, padded V for WMMA tiles

#define NGRP_COLS 208                // 13 n-tiles per n-group; 5 * 208 == VPAD
#define SLAB_HALF (NGRP_COLS * JH)   // 133,120 f16 elements
#define SLAB_BYTES (SLAB_HALF * 2)   // 266,240 B  (< 320 KB WGP LDS)
#define SLAB_SEGS (SLAB_BYTES / 16)  // 16,640 x b128 segments
#define WAVES_PER_BLOCK 16           // 512 threads; 16 m-tiles per block

typedef _Float16 v16h __attribute__((ext_vector_type(16)));
typedef float    v8f  __attribute__((ext_vector_type(8)));
typedef float    v4f  __attribute__((ext_vector_type(4)));
typedef int      v4i  __attribute__((ext_vector_type(4)));

#define GAS __attribute__((address_space(1)))   // global
#define LAS __attribute__((address_space(3)))   // LDS

#if __has_builtin(__builtin_amdgcn_global_load_async_to_lds_b128)
#define HAVE_ASYNC_LDS 1
#else
#define HAVE_ASYNC_LDS 0
#endif

__device__ __forceinline__ void wait_asynccnt0() {
#if __has_builtin(__builtin_amdgcn_s_wait_asynccnt)
  __builtin_amdgcn_s_wait_asynccnt(0);
#else
  asm volatile("s_wait_asynccnt 0x0" ::: "memory");
#endif
}

// ---------------- conversion kernels ----------------
__global__ void cvt_f16_kernel(const float* __restrict__ src,
                               _Float16* __restrict__ dst, int n) {
  int i = blockIdx.x * blockDim.x + threadIdx.x;
  if (i < n) dst[i] = (_Float16)src[i];
}

// out_w [1025,640] f32 -> f16 padded to [1040,640] (pad rows = 0)
__global__ void cvt_outw_kernel(const float* __restrict__ src,
                                _Float16* __restrict__ dst) {
  int i = blockIdx.x * blockDim.x + threadIdx.x;
  if (i >= VPAD * JH) return;
  int row = i / JH;
  dst[i] = (row < VV) ? (_Float16)src[i] : (_Float16)0.0f;
}

// ---------------- f = enc16 @ enc_w16^T + enc_b  (f32 out) ----------------
// One wave per 16(m) x 128(n) tile. A,W both row-major [rows, K] in f16.
__global__ __launch_bounds__(32)
void enc_proj_wmma(const _Float16* __restrict__ A,   // [B*T, EH]
                   const _Float16* __restrict__ W,   // [JH, EH]
                   const float* __restrict__ bias,   // [JH]
                   float* __restrict__ F)            // [B*T, JH]
{
  const int lane = threadIdx.x;
  const int m0   = blockIdx.x * 16;
  const int n0   = blockIdx.y * 128;      // 8 n-tiles
  const int rlo  = lane & 15;
  const int kh   = (lane >> 4) * 8;       // K half-select per A/B f16 layout

  v8f acc[8] = {};

  const _Float16* arow = A + (size_t)(m0 + rlo) * EH + kh;
  for (int k0 = 0; k0 < EH; k0 += 32) {
    v16h a;
    v4f* ap = (v4f*)&a;
    ap[0] = *(const v4f*)(arow + k0);          // K = k0+kh .. +7
    ap[1] = *(const v4f*)(arow + k0 + 16);     // K = k0+16+kh .. +7
#pragma unroll
    for (int nt = 0; nt < 8; ++nt) {
      const _Float16* wrow = W + (size_t)(n0 + nt * 16 + rlo) * EH + kh + k0;
      v16h b;
      v4f* bp = (v4f*)&b;
      bp[0] = *(const v4f*)(wrow);
      bp[1] = *(const v4f*)(wrow + 16);
      acc[nt] = __builtin_amdgcn_wmma_f32_16x16x32_f16(
          false, a, false, b, (short)0, acc[nt], false, false);
    }
  }

  const int mhi = (lane >> 4) * 8;          // D layout: VGPR r -> M=r / M=8+r
#pragma unroll
  for (int nt = 0; nt < 8; ++nt) {
    int n = n0 + nt * 16 + rlo;
    float bv = bias[n];
#pragma unroll
    for (int r = 0; r < 8; ++r)
      F[(size_t)(m0 + mhi + r) * JH + n] = acc[nt][r] + bv;
  }
}

// ---------------- g = transpose(dec) @ pred_w^T + pred_b (tiny, VALU) ------
__global__ void pred_proj(const float* __restrict__ dec,   // [B, PH, U]
                          const float* __restrict__ W,     // [JH, PH]
                          const float* __restrict__ bias,  // [JH]
                          float* __restrict__ G)           // [B*U, JH]
{
  int tid = blockIdx.x * blockDim.x + threadIdx.x;
  if (tid >= BB * UU * JH) return;
  int j = tid % JH;
  int m = tid / JH;
  int b = m / UU, u = m % UU;
  const float* dcol = dec + (size_t)b * PH * UU + u;  // stride UU over h
  const float* wrow = W + (size_t)j * PH;
  float s = bias[j];
  for (int h = 0; h < PH; ++h) s += dcol[h * UU] * wrow[h];
  G[tid] = s;
}

// ---------------- main: res = relu(f + g) @ out_w16^T + out_b --------------
// 16 waves per block share one 266 KB LDS slab of out_w16 columns
// (async-filled once). Each wave owns a 16(u) x 208(v) tile: 13 n-tile
// accumulators, B fragments from LDS, A built on the fly in f32 with the
// next k-step's f/g loads software-pipelined past the WMMA burst.
__global__ __launch_bounds__(WAVES_PER_BLOCK * 32)
void joint_wmma(const float* __restrict__ F,       // [B*T, JH]
                const float* __restrict__ G,       // [B*U, JH]
                const _Float16* __restrict__ W,    // [VPAD, JH] f16
                const float* __restrict__ ob,      // [V]
                float* __restrict__ out)           // [B*T*U, V]
{
  __shared__ _Float16 sW[SLAB_HALF];   // 266,240 B (fits 320 KB WGP LDS)

  const int lane = threadIdx.x & 31;
  const int wave = threadIdx.x >> 5;
  const int n0   = blockIdx.y * NGRP_COLS;

  // ---- cooperative slab fill: out_w16 rows n0..n0+207 (contiguous) ----
  {
    const char* gsrc = (const char*)(W + (size_t)n0 * JH);
    char* lbase = (char*)sW;
    for (int s = threadIdx.x; s < SLAB_SEGS; s += (int)blockDim.x) {
#if HAVE_ASYNC_LDS
      __builtin_amdgcn_global_load_async_to_lds_b128(
          (GAS v4i*)(gsrc + 16 * s),
          (LAS v4i*)(lbase + 16 * s),
          0, 0);
#else
      *(v4f*)(lbase + 16 * s) = *(const v4f*)(gsrc + 16 * s);
#endif
    }
#if HAVE_ASYNC_LDS
    wait_asynccnt0();
#endif
  }
  __syncthreads();

  // ---- per-wave m-tile: (b, t, u-tile) ----
  int mt = blockIdx.x * WAVES_PER_BLOCK + wave;   // 0 .. 6143
  const int ut = mt % (UU / 16); mt /= (UU / 16);
  const int t  = mt % TT;
  const int b  = mt / TT;
  const int u0 = ut * 16;
  const int rlo = lane & 15;
  const int kh  = (lane >> 4) * 8;

  const float* frow = F + (size_t)(b * TT + t) * JH + kh;        // shared row
  const float* grow = G + (size_t)(b * UU + u0 + rlo) * JH + kh; // per-u row
  const _Float16* lw = sW + (size_t)rlo * JH + kh;               // lane's B row

  v8f acc[13] = {};

  // prologue: load k-step 0 operands
  v4f f0 = *(const v4f*)(frow + 0),  f1 = *(const v4f*)(frow + 4);
  v4f f2 = *(const v4f*)(frow + 16), f3 = *(const v4f*)(frow + 20);
  v4f g0 = *(const v4f*)(grow + 0),  g1 = *(const v4f*)(grow + 4);
  v4f g2 = *(const v4f*)(grow + 16), g3 = *(const v4f*)(grow + 20);

  for (int k0 = 0; k0 < JH; k0 += 32) {
    // A fragment: relu(f+g) in f32, quantize to f16 per WMMA A layout
    v16h a;
#pragma unroll
    for (int i = 0; i < 4; ++i) {
      a[i]      = (_Float16)fmaxf(f0[i] + g0[i], 0.0f);
      a[4 + i]  = (_Float16)fmaxf(f1[i] + g1[i], 0.0f);
      a[8 + i]  = (_Float16)fmaxf(f2[i] + g2[i], 0.0f);
      a[12 + i] = (_Float16)fmaxf(f3[i] + g3[i], 0.0f);
    }

    // branchless prefetch of next k-step's f/g (wraps to 0 on last iter)
    const int kn = (k0 + 32 < JH) ? (k0 + 32) : 0;
    f0 = *(const v4f*)(frow + kn);      f1 = *(const v4f*)(frow + kn + 4);
    f2 = *(const v4f*)(frow + kn + 16); f3 = *(const v4f*)(frow + kn + 20);
    g0 = *(const v4f*)(grow + kn);      g1 = *(const v4f*)(grow + kn + 4);
    g2 = *(const v4f*)(grow + kn + 16); g3 = *(const v4f*)(grow + kn + 20);

#pragma unroll
    for (int nt = 0; nt < 13; ++nt) {
      const _Float16* wrow = lw + (size_t)(nt * 16) * JH + k0;   // ds_load path
      v16h bm;
      v4f* bp = (v4f*)&bm;
      bp[0] = *(const v4f*)(wrow);
      bp[1] = *(const v4f*)(wrow + 16);
      acc[nt] = __builtin_amdgcn_wmma_f32_16x16x32_f16(
          false, a, false, bm, (short)0, acc[nt], false, false);
    }
  }

  const int mhi = (lane >> 4) * 8;          // D layout: VGPR r -> M=r / M=8+r
  const size_t rowbase = (size_t)((b * TT + t) * UU + u0 + mhi);
#pragma unroll
  for (int nt = 0; nt < 13; ++nt) {
    int n = n0 + nt * 16 + rlo;
    if (n < VV) {
      float bv = ob[n];
#pragma unroll
      for (int r = 0; r < 8; ++r) {
        // streaming output (403 MB, write-once): non-temporal to spare L2
        __builtin_nontemporal_store(acc[nt][r] + bv,
                                    &out[(rowbase + r) * (size_t)VV + n]);
      }
    }
  }
}

// ---------------- host launcher ----------------
extern "C" void kernel_launch(void* const* d_in, const int* in_sizes, int n_in,
                              void* d_out, int out_size, void* d_ws, size_t ws_size,
                              hipStream_t stream) {
  const float* enc    = (const float*)d_in[0];
  const float* dec    = (const float*)d_in[1];
  const float* enc_w  = (const float*)d_in[2];
  const float* enc_b  = (const float*)d_in[3];
  const float* pred_w = (const float*)d_in[4];
  const float* pred_b = (const float*)d_in[5];
  const float* out_w  = (const float*)d_in[6];
  const float* out_b  = (const float*)d_in[7];
  float* out = (float*)d_out;

  char* ws = (char*)d_ws;
  // workspace layout (256B aligned regions, ~8.4 MB total)
  _Float16* enc16  = (_Float16*)(ws + 0);          // 1024*1024*2 = 2,097,152
  _Float16* encw16 = (_Float16*)(ws + 2097152);    //  640*1024*2 = 1,310,720
  _Float16* outw16 = (_Float16*)(ws + 3407872);    // 1040* 640*2 = 1,331,200
  float*    F      = (float*)   (ws + 4739072);    // 1024* 640*4 = 2,621,440
  float*    G      = (float*)   (ws + 7360512);    //  384* 640*4 =   983,040

  cvt_f16_kernel<<<(BB * TT * EH + 255) / 256, 256, 0, stream>>>(enc, enc16, BB * TT * EH);
  cvt_f16_kernel<<<(JH * EH + 255) / 256, 256, 0, stream>>>(enc_w, encw16, JH * EH);
  cvt_outw_kernel<<<(VPAD * JH + 255) / 256, 256, 0, stream>>>(out_w, outw16);

  enc_proj_wmma<<<dim3(BB * TT / 16, JH / 128), 32, 0, stream>>>(enc16, encw16, enc_b, F);
  pred_proj<<<(BB * UU * JH + 255) / 256, 256, 0, stream>>>(dec, pred_w, pred_b, G);

  joint_wmma<<<dim3(BB * TT * (UU / 16) / WAVES_PER_BLOCK, 5),
               WAVES_PER_BLOCK * 32, 0, stream>>>(F, G, outw16, out_b, out);
}
